// ManualLSTM_70506183131641
// MI455X (gfx1250) — matrix operand order repaired and checked by reading, not compiled, not run on hardware
//
#include <hip/hip_runtime.h>
#include <stddef.h>

// ---------------- problem constants ----------------
#define BATCH   32
#define SEQLEN  1024
#define INDIM   512
#define HID     512
#define G4      2048           // 4*HID, gate order [i,f,o,g]
#define MTOT    (BATCH*SEQLEN) // 32768 rows of x

// scan decomposition
#define NWG     32             // persistent workgroups
#define HSL     16             // hidden units per workgroup (NWG*HSL == HID)

// ---------------- workspace layout (bytes) ----------------
#define WS_BAR   0                                   // 2 uints (barrier)
#define WS_BIAS  256                                 // fp32[2048]
#define WS_WIT   (WS_BIAS + 8192)                    // bf16 WiT [2048][512]
#define WS_WHT   (WS_WIT  + 2097152)                 // bf16 WhT [2048][512]
#define WS_H     (WS_WHT  + 2097152)                 // bf16 h ping-pong 2*[32][512]
#define WS_XBF   (WS_H    + 65536)                   // bf16 x [32768][512]
#define WS_XPRE  (WS_XBF  + 33554432)                // fp32 x_pre [t][b][2048]

#define OUT_H    16777216                            // outputs come first
#define OUT_C    (OUT_H + 16384)

typedef __attribute__((ext_vector_type(16))) __bf16        v16bf;
typedef __attribute__((ext_vector_type(8)))  float         v8f;
typedef __attribute__((ext_vector_type(4)))  unsigned int  uint4v;

union bf16x16 { uint4v u[2]; v16bf v; };

__device__ __forceinline__ unsigned short f2bf(float f) {
  unsigned int u = __float_as_uint(f);
  u = u + 0x7FFFu + ((u >> 16) & 1u);      // round-to-nearest-even
  return (unsigned short)(u >> 16);
}
__device__ __forceinline__ float sigm(float x) { return 1.0f / (1.0f + __expf(-x)); }
__device__ __forceinline__ float ftanh(float x) {
  float e = __expf(-2.0f * fabsf(x));
  float r = (1.0f - e) / (1.0f + e);
  return copysignf(r, x);
}

// ---------------- pack: bias concat, h0->bf16, barrier reset ----------------
__global__ void k_init(const float* __restrict__ h0,
                       const float* __restrict__ bi, const float* __restrict__ bf,
                       const float* __restrict__ bo, const float* __restrict__ bg,
                       float* __restrict__ bias, unsigned short* __restrict__ hbuf,
                       unsigned int* __restrict__ bar) {
  int gid = blockIdx.x * 256 + threadIdx.x;
  if (gid < 2) bar[gid] = 0u;
  if (gid < G4) {
    int q = gid >> 9, j = gid & 511;
    const float* src = (q == 0) ? bi : (q == 1) ? bf : (q == 2) ? bo : bg;
    bias[gid] = src[j];
  } else if (gid < G4 + BATCH * HID) {
    int e = gid - G4;
    hbuf[e] = f2bf(h0[e]);                 // ping buffer 0 = h_0
  }
}

// ---------------- pack weights: fp32 [K][N-slice] -> bf16 transposed [4H][K] ----------------
__global__ void k_pack_w(const float* __restrict__ Wii, const float* __restrict__ Wif,
                         const float* __restrict__ Wio, const float* __restrict__ Wig,
                         const float* __restrict__ Whi, const float* __restrict__ Whf,
                         const float* __restrict__ Who, const float* __restrict__ Whg,
                         unsigned short* __restrict__ wit, unsigned short* __restrict__ wht) {
  int e = blockIdx.x * 256 + threadIdx.x;          // 0 .. 2*2048*512-1
  int which = (e >= G4 * INDIM) ? 1 : 0;
  int idx = which ? e - G4 * INDIM : e;
  int n = idx >> 9;                                // row of transposed weight (gate-major col)
  int k = idx & 511;
  int q = n >> 9, j = n & 511;
  const float* src;
  if (!which) src = (q == 0) ? Wii : (q == 1) ? Wif : (q == 2) ? Wio : Wig;
  else        src = (q == 0) ? Whi : (q == 1) ? Whf : (q == 2) ? Who : Whg;
  unsigned short v = f2bf(src[(size_t)k * HID + j]);
  (which ? wht : wit)[idx] = v;
}

// ---------------- pack x: fp32 -> bf16, 4 elems/thread ----------------
__global__ void k_pack_x(const float* __restrict__ x, unsigned short* __restrict__ xbf) {
  size_t gid = (size_t)blockIdx.x * 256 + threadIdx.x;   // 4,194,304 threads
  const float4 v = ((const float4*)x)[gid];
  unsigned int lo = (unsigned int)f2bf(v.x) | ((unsigned int)f2bf(v.y) << 16);
  unsigned int hi = (unsigned int)f2bf(v.z) | ((unsigned int)f2bf(v.w) << 16);
  ((unsigned int*)xbf)[gid * 2]     = lo;
  ((unsigned int*)xbf)[gid * 2 + 1] = hi;
}

// ---------------- input GEMM: x_pre[t][b][g] = x @ Wi + b ----------------
// Each wave computes a 32x64 tile: 2 A tiles x 4 B tiles, K-loop of 16 wmma steps.
__global__ __launch_bounds__(256, 2)
void k_input_gemm(const unsigned short* __restrict__ xbf,
                  const unsigned short* __restrict__ wit,
                  const float* __restrict__ bias,
                  float* __restrict__ xpre) {
  const int w     = (blockIdx.x * 256 + threadIdx.x) >> 5;   // 0..32767
  const int lane  = threadIdx.x & 31;
  const int ln16  = lane & 15;
  const int hi    = lane >> 4;
  const int m0    = (w >> 5) * 32;       // 1024 M tiles
  const int n0    = (w & 31) * 64;       // 32 N tiles

  v8f acc[2][4];
  #pragma unroll
  for (int mt = 0; mt < 2; ++mt)
    #pragma unroll
    for (int nt = 0; nt < 4; ++nt) acc[mt][nt] = v8f{};

  #pragma unroll 4
  for (int kk = 0; kk < 16; ++kk) {
    // A: 16-bit 16x32 layout — lane<16 holds K{0..7,16..23}, lane>=16 K{8..15,24..31}
    const int kbA = kk * 32 + hi * 8;
    bf16x16 a[2];
    #pragma unroll
    for (int mt = 0; mt < 2; ++mt) {
      const unsigned short* p = xbf + (size_t)(m0 + mt * 16 + ln16) * INDIM + kbA;
      a[mt].u[0] = *(const uint4v*)(p);
      a[mt].u[1] = *(const uint4v*)(p + 16);
    }
    // B: lanes 0-15 hold K{0..7,16..23}, lanes 16-31 hold K{8..15,24..31} (from WiT[N][K], row-major)
    const int kbB = kk * 32 + hi * 8;
    #pragma unroll
    for (int nt = 0; nt < 4; ++nt) {
      bf16x16 b;
      const unsigned short* p = wit + (size_t)(n0 + nt * 16 + ln16) * INDIM + kbB;
      b.u[0] = *(const uint4v*)(p);
      b.u[1] = *(const uint4v*)(p + 16);
      acc[0][nt] = __builtin_amdgcn_wmma_f32_16x16x32_bf16(false, a[0].v, false, b.v,
                                                           (short)0, acc[0][nt], false, false);
      acc[1][nt] = __builtin_amdgcn_wmma_f32_16x16x32_bf16(false, a[1].v, false, b.v,
                                                           (short)0, acc[1][nt], false, false);
    }
  }
  // store with bias into x_pre laid out [t][b][2048] (scan-friendly)
  #pragma unroll
  for (int nt = 0; nt < 4; ++nt) {
    const int n = n0 + nt * 16 + ln16;
    const float bv = bias[n];
    #pragma unroll
    for (int mt = 0; mt < 2; ++mt) {
      #pragma unroll
      for (int r = 0; r < 8; ++r) {
        int m = m0 + mt * 16 + r + hi * 8;       // m = b*1024 + t
        int b = m >> 10, t = m & 1023;
        xpre[((size_t)t * BATCH + b) * G4 + n] = acc[mt][nt][r] + bv;
      }
    }
  }
}

// ---------------- persistent recurrent scan ----------------
// 32 workgroups x 128 threads. Workgroup g owns hidden units [g*16, g*16+16).
// Wave q (0..3) owns gate q; its 16x512 bf16 slice of Wh is pinned in 128 VGPRs.
__global__ __launch_bounds__(128, 1)
void k_scan(const float* __restrict__ xpre, const unsigned short* __restrict__ wht,
            unsigned short* __restrict__ hbuf, const float* __restrict__ c0,
            float* __restrict__ out, unsigned int* __restrict__ bar) {
  const int g    = blockIdx.x;
  const int tid  = threadIdx.x;
  const int q    = tid >> 5;
  const int lane = tid & 31;
  const int ln16 = lane & 15;
  const int hi   = lane >> 4;
  const int j0   = g * HSL;

  __shared__ float lds_g[4][BATCH][HSL];
  __shared__ float lds_c[BATCH][HSL];
  __shared__ float lds_h[BATCH][HSL];

  for (int e = tid; e < BATCH * HSL; e += 128) {
    int b = e >> 4, j = e & 15;
    lds_c[b][j] = c0[(size_t)b * HID + j0 + j];
  }

  // pin this wave's Wh gate-slice in registers for all 1024 steps
  v16bf Breg[16];
  {
    const unsigned short* base = wht + (size_t)(q * HID + j0 + ln16) * HID;
    #pragma unroll
    for (int kk = 0; kk < 16; ++kk) {
      bf16x16 tmp;
      const unsigned short* p = base + kk * 32 + hi * 8;
      tmp.u[0] = *(const uint4v*)(p);
      tmp.u[1] = *(const uint4v*)(p + 16);
      Breg[kk] = tmp.v;
    }
  }
  __syncthreads();

  for (int t = 0; t < SEQLEN; ++t) {
    const unsigned short* hr = hbuf + (size_t)(t & 1) * (BATCH * HID);        // read
    unsigned short*       hw = hbuf + (size_t)((t & 1) ^ 1) * (BATCH * HID);  // write

    v8f acc0 = v8f{}, acc1 = v8f{};
    #pragma unroll
    for (int kk = 0; kk < 16; ++kk) {
      const int kb = kk * 32 + hi * 8;
      bf16x16 a0, a1;
      const unsigned short* r0 = hr + (size_t)ln16 * HID + kb;
      const unsigned short* r1 = hr + (size_t)(16 + ln16) * HID + kb;
      a0.u[0] = *(const uint4v*)(r0);  a0.u[1] = *(const uint4v*)(r0 + 16);
      a1.u[0] = *(const uint4v*)(r1);  a1.u[1] = *(const uint4v*)(r1 + 16);
      acc0 = __builtin_amdgcn_wmma_f32_16x16x32_bf16(false, a0.v, false, Breg[kk],
                                                     (short)0, acc0, false, false);
      acc1 = __builtin_amdgcn_wmma_f32_16x16x32_bf16(false, a1.v, false, Breg[kk],
                                                     (short)0, acc1, false, false);
    }

    // gates = h@Wh + x_pre  -> LDS (C/D layout: vgpr r, lane -> M = r + hi*8, N = ln16)
    const float* xp = xpre + (size_t)t * (BATCH * G4) + q * HID + j0 + ln16;
    #pragma unroll
    for (int r = 0; r < 8; ++r) {
      int b0 = r + hi * 8;
      lds_g[q][b0][ln16]      = acc0[r] + xp[(size_t)b0 * G4];
      lds_g[q][b0 + 16][ln16] = acc1[r] + xp[(size_t)(b0 + 16) * G4];
    }
    if (t + 1 < SEQLEN)   // pull next step's activations toward L2/L0
      __builtin_prefetch(xpre + (size_t)(t + 1) * (BATCH * G4) + q * HID + j0 + (lane << 3), 0, 1);
    __syncthreads();

    for (int e = tid; e < BATCH * HSL; e += 128) {
      int b = e >> 4, j = e & 15;
      float iv = sigm(lds_g[0][b][j]);
      float fv = sigm(lds_g[1][b][j]);
      float ov = sigm(lds_g[2][b][j]);
      float gv = ftanh(lds_g[3][b][j]);
      float c  = fv * lds_c[b][j] + iv * gv;
      float h  = ov * ftanh(c);
      lds_c[b][j] = c;
      lds_h[b][j] = h;
      out[((size_t)b * SEQLEN + t) * HID + j0 + j] = h;
      hw[(size_t)b * HID + j0 + j] = f2bf(h);
    }
    __syncthreads();

    // grid barrier (sense-reversal) with agent-scope fences
    if (tid == 0) {
      __builtin_amdgcn_fence(__ATOMIC_RELEASE, "agent");
      unsigned int gen = __hip_atomic_load(&bar[1], __ATOMIC_RELAXED, __HIP_MEMORY_SCOPE_AGENT);
      unsigned int arr = __hip_atomic_fetch_add(&bar[0], 1u, __ATOMIC_ACQ_REL, __HIP_MEMORY_SCOPE_AGENT);
      if (arr == NWG - 1) {
        __hip_atomic_store(&bar[0], 0u, __ATOMIC_RELAXED, __HIP_MEMORY_SCOPE_AGENT);
        __hip_atomic_fetch_add(&bar[1], 1u, __ATOMIC_RELEASE, __HIP_MEMORY_SCOPE_AGENT);
      } else {
        while (__hip_atomic_load(&bar[1], __ATOMIC_ACQUIRE, __HIP_MEMORY_SCOPE_AGENT) == gen)
          __builtin_amdgcn_s_sleep(1);
      }
      __builtin_amdgcn_fence(__ATOMIC_ACQUIRE, "agent");
    }
    __syncthreads();
  }

  for (int e = tid; e < BATCH * HSL; e += 128) {
    int b = e >> 4, j = e & 15;
    out[OUT_H + (size_t)b * HID + j0 + j] = lds_h[b][j];
    out[OUT_C + (size_t)b * HID + j0 + j] = lds_c[b][j];
  }
}

// ---------------- launch ----------------
extern "C" void kernel_launch(void* const* d_in, const int* in_sizes, int n_in,
                              void* d_out, int out_size, void* d_ws, size_t ws_size,
                              hipStream_t stream) {
  (void)in_sizes; (void)n_in; (void)out_size; (void)ws_size;
  const float* x   = (const float*)d_in[0];
  const float* h0  = (const float*)d_in[1];
  const float* c0  = (const float*)d_in[2];
  const float* Wii = (const float*)d_in[3];
  const float* Wif = (const float*)d_in[4];
  const float* Wio = (const float*)d_in[5];
  const float* Wig = (const float*)d_in[6];
  const float* Whi = (const float*)d_in[7];
  const float* Whf = (const float*)d_in[8];
  const float* Who = (const float*)d_in[9];
  const float* Whg = (const float*)d_in[10];
  const float* bi  = (const float*)d_in[11];
  const float* bf  = (const float*)d_in[12];
  const float* bo  = (const float*)d_in[13];
  const float* bg  = (const float*)d_in[14];

  char* ws = (char*)d_ws;
  unsigned int*   bar  = (unsigned int*)(ws + WS_BAR);
  float*          bias = (float*)(ws + WS_BIAS);
  unsigned short* wit  = (unsigned short*)(ws + WS_WIT);
  unsigned short* wht  = (unsigned short*)(ws + WS_WHT);
  unsigned short* hbuf = (unsigned short*)(ws + WS_H);
  unsigned short* xbf  = (unsigned short*)(ws + WS_XBF);
  float*          xpre = (float*)(ws + WS_XPRE);

  k_init<<<72, 256, 0, stream>>>(h0, bi, bf, bo, bg, bias, hbuf, bar);
  k_pack_w<<<8192, 256, 0, stream>>>(Wii, Wif, Wio, Wig, Whi, Whf, Who, Whg, wit, wht);
  k_pack_x<<<16384, 256, 0, stream>>>(x, xbf);
  k_input_gemm<<<4096, 256, 0, stream>>>(xbf, wit, bias, xpre);
  k_scan<<<NWG, 128, 0, stream>>>(xpre, wht, hbuf, c0, (float*)d_out, bar);
}
